// VAE_79104707658332
// MI455X (gfx1250) — compile-verified, matrix operand order checked by reading, not run on hardware
//
#include <hip/hip_runtime.h>
#include <hip/hip_bf16.h>
#include <math.h>

typedef __attribute__((ext_vector_type(16))) _Float16 v16h;
typedef __attribute__((ext_vector_type(8)))  _Float16 v8h;
typedef __attribute__((ext_vector_type(8)))  float    v8f;

#define ENC_LDA 520   // 512 + 8 halves pad: row stride 260 dwords -> 4-bank stagger
#define DEC_LDA 136   // 128 + 8 halves pad
#define NT_LDA  136   // W_nt LDS row stride (K=128 + 8 pad)

// ---------------------------------------------------------------------------
// Fragment loader for v_wmma_f32_16x16x32_f16.
// 16-bit A-matrix 16x32 layout (ISA 7.12.2): lane<16 -> row M=lane, K chunks
// {0..7} (elems 0..7) and {16..23} (elems 8..15); lane>=16 -> row M=lane-16,
// K {8..15} and {24..31}.  Caller passes p = rowBase + kb, kb=(lane&16)?8:0.
// B fragments use the mirrored layout (lane -> N column).
// ---------------------------------------------------------------------------
__device__ __forceinline__ v16h make_frag(const _Float16* p) {
    v8h lo = *(const v8h*)(p);
    v8h hi = *(const v8h*)(p + 16);
    v16h f;
#pragma unroll
    for (int i = 0; i < 8; ++i) { f[i] = lo[i]; f[i + 8] = hi[i]; }
    return f;
}

#define WMMA(A, Bm, C) __builtin_amdgcn_wmma_f32_16x16x32_f16(false, (A), false, (Bm), (short)0, (C), false, false)

// ---------------------------------------------------------------------------
__global__ __launch_bounds__(256)
void f32_to_f16_kernel(const float* __restrict__ src, _Float16* __restrict__ dst, int n) {
    int i = blockIdx.x * 256 + threadIdx.x;
    if (i < n) dst[i] = (_Float16)src[i];
}

// h0[b,l,:] = tanh(embed[tok[b,l],:]) -> fp16, 4 elems per thread
__global__ __launch_bounds__(256)
void embed_kernel(const int* __restrict__ tok, const float* __restrict__ emb,
                  _Float16* __restrict__ h0) {
    int idx = blockIdx.x * 256 + threadIdx.x;   // over B*L*(H/4)
    int hq  = idx & 63;
    int bl  = idx >> 6;
    int t   = tok[bl];
    float4 e = ((const float4*)(emb + (size_t)t * 256))[hq];
    _Float16* dst = h0 + (size_t)bl * 256 + hq * 4;
    dst[0] = (_Float16)tanhf(e.x);
    dst[1] = (_Float16)tanhf(e.y);
    dst[2] = (_Float16)tanhf(e.z);
    dst[3] = (_Float16)tanhf(e.w);
}

// ---------------------------------------------------------------------------
// Encoder level: A = hin viewed as (B*np, 512) fp16 (pairing == reinterpret).
// Block = 512 threads (16 waves) = 32 rows x 256 cols.  Wave w owns column
// tile w and BOTH 16-row tiles, so each 3-op B-fragment set is reused for two
// row tiles (halves weight L2 traffic vs 16-row blocks).  A staged in padded
// LDS.  Per-row op select + bias + tanh in epilogue.
// ---------------------------------------------------------------------------
__global__ __launch_bounds__(512)
void enc_level_kernel(const _Float16* __restrict__ hin,
                      const _Float16* __restrict__ Wenc,   // (3,256,512) f16
                      const float*    __restrict__ benc,   // (3,256)
                      const int*      __restrict__ ops,    // (B,127)
                      _Float16*       __restrict__ hout,   // (B*np, 256)
                      int np_log2, int level_off) {
    __shared__ _Float16 As[32 * ENC_LDA];
    const int tid  = threadIdx.x;
    const int lane = tid & 31;
    const int wid  = tid >> 5;              // 0..15 -> column tile
    const int m0   = blockIdx.x * 32;

    // cooperative A-tile load: 32 rows x 64 uint4, 4 per thread
#pragma unroll
    for (int i = 0; i < 4; ++i) {
        int vi = tid + i * 512;
        int rr = vi >> 6, cc = vi & 63;
        ((uint4*)(As + rr * ENC_LDA))[cc] =
            ((const uint4*)(hin + (size_t)(m0 + rr) * 512))[cc];
    }
    __syncthreads();

    const int r    = lane & 15;
    const int kb   = (lane & 16) ? 8 : 0;
    const int hi8  = (lane & 16) ? 8 : 0;
    const int npm  = (1 << np_log2) - 1;
    const int ncol = wid * 16 + r;

    const _Float16* w0 = Wenc + (size_t)ncol * 512 + kb;
    const _Float16* w1 = w0 + 256 * 512;
    const _Float16* w2 = w1 + 256 * 512;
    const _Float16* a0 = As + r * ENC_LDA + kb;
    const _Float16* a1 = As + (16 + r) * ENC_LDA + kb;

    v8f acc[2][3] = {};
#pragma unroll 2
    for (int kk = 0; kk < 512; kk += 32) {
        v16h b0 = make_frag(w0 + kk);
        v16h b1 = make_frag(w1 + kk);
        v16h b2 = make_frag(w2 + kk);
        v16h aA = make_frag(a0 + kk);
        v16h aB = make_frag(a1 + kk);
        acc[0][0] = WMMA(aA, b0, acc[0][0]);
        acc[0][1] = WMMA(aA, b1, acc[0][1]);
        acc[0][2] = WMMA(aA, b2, acc[0][2]);
        acc[1][0] = WMMA(aB, b0, acc[1][0]);
        acc[1][1] = WMMA(aB, b1, acc[1][1]);
        acc[1][2] = WMMA(aB, b2, acc[1][2]);
    }

#pragma unroll
    for (int rt = 0; rt < 2; ++rt) {
#pragma unroll
        for (int v = 0; v < 8; ++v) {
            int m  = m0 + rt * 16 + v + hi8;
            int b  = m >> np_log2;
            int j  = m & npm;
            int op = ops[b * 127 + level_off + j];
            float y = (op == 0) ? acc[rt][0][v] : ((op == 1) ? acc[rt][1][v] : acc[rt][2][v]);
            y += benc[op * 256 + ncol];
            hout[(size_t)m * 256 + ncol] = (_Float16)tanhf(y);
        }
    }
}

// ---------------------------------------------------------------------------
// Latent head (0.13 GFLOP, VALU): mu/logvar fp32; z -> codes[b,0,:] fp16.
// ---------------------------------------------------------------------------
__global__ __launch_bounds__(256)
void latent_kernel(const _Float16* __restrict__ hroot,
                   const float* __restrict__ Wmu, const float* __restrict__ bmu,
                   const float* __restrict__ Wlv, const float* __restrict__ blv,
                   const float* __restrict__ eps,
                   float* __restrict__ mu_out, float* __restrict__ lv_out,
                   _Float16* __restrict__ codes) {
    int idx = blockIdx.x * 256 + threadIdx.x;    // B*128
    int b = idx >> 7, l = idx & 127;
    const _Float16* h = hroot + (size_t)b * 256;
    const float* wm = Wmu + (size_t)l * 256;
    const float* wl = Wlv + (size_t)l * 256;
    float mu = bmu[l], lv = blv[l];
    for (int i = 0; i < 256; ++i) {
        float hv = (float)h[i];
        mu += hv * wm[i];
        lv += hv * wl[i];
    }
    mu_out[idx] = mu;
    lv_out[idx] = lv;
    float z = mu + eps[idx] * expf(0.5f * lv);
    codes[(size_t)b * 255 * 128 + l] = (_Float16)z;
}

// ---------------------------------------------------------------------------
// Decoder level: A rows = codes[b, inoff+j, :] (K=128).  Same 32-row/512-
// thread structure as the encoder; output 256 cols per node written straight
// into codes[b, outoff+2j(+1), :] (the lc/rc interleave is a reinterpret).
// ---------------------------------------------------------------------------
__global__ __launch_bounds__(512)
void dec_level_kernel(const _Float16* __restrict__ codes_in,
                      const _Float16* __restrict__ Wdec,   // (3,256,128) f16
                      const float*    __restrict__ bdec,   // (3,256)
                      const int*      __restrict__ ops,
                      _Float16*       __restrict__ codes_out,
                      int n_log2, int inoff, int outoff, int level_off) {
    __shared__ _Float16 As[32 * DEC_LDA];
    const int tid  = threadIdx.x;
    const int lane = tid & 31;
    const int wid  = tid >> 5;
    const int m0   = blockIdx.x * 32;
    const int nm   = (1 << n_log2) - 1;

    // cooperative A-tile load: 32 rows x 16 uint4 (per-row gather), 1/thread
    {
        int rr = tid >> 4, cc = tid & 15;
        int m  = m0 + rr;
        int b  = m >> n_log2;
        int j  = m & nm;
        ((uint4*)(As + rr * DEC_LDA))[cc] =
            ((const uint4*)(codes_in + ((size_t)b * 255 + inoff + j) * 128))[cc];
    }
    __syncthreads();

    const int r    = lane & 15;
    const int kb   = (lane & 16) ? 8 : 0;
    const int hi8  = (lane & 16) ? 8 : 0;
    const int ncol = wid * 16 + r;

    const _Float16* w0 = Wdec + (size_t)ncol * 128 + kb;
    const _Float16* w1 = w0 + 256 * 128;
    const _Float16* w2 = w1 + 256 * 128;
    const _Float16* a0 = As + r * DEC_LDA + kb;
    const _Float16* a1 = As + (16 + r) * DEC_LDA + kb;

    v8f acc[2][3] = {};
#pragma unroll
    for (int kk = 0; kk < 128; kk += 32) {
        v16h b0 = make_frag(w0 + kk);
        v16h b1 = make_frag(w1 + kk);
        v16h b2 = make_frag(w2 + kk);
        v16h aA = make_frag(a0 + kk);
        v16h aB = make_frag(a1 + kk);
        acc[0][0] = WMMA(aA, b0, acc[0][0]);
        acc[0][1] = WMMA(aA, b1, acc[0][1]);
        acc[0][2] = WMMA(aA, b2, acc[0][2]);
        acc[1][0] = WMMA(aB, b0, acc[1][0]);
        acc[1][1] = WMMA(aB, b1, acc[1][1]);
        acc[1][2] = WMMA(aB, b2, acc[1][2]);
    }

#pragma unroll
    for (int rt = 0; rt < 2; ++rt) {
#pragma unroll
        for (int v = 0; v < 8; ++v) {
            int m  = m0 + rt * 16 + v + hi8;
            int b  = m >> n_log2;
            int j  = m & nm;
            int op = ops[b * 127 + level_off + j];
            float y = (op == 0) ? acc[rt][0][v] : ((op == 1) ? acc[rt][1][v] : acc[rt][2][v]);
            y = tanhf(y + bdec[op * 256 + ncol]);
            codes_out[((size_t)b * 255 + outoff + 2 * j) * 128 + ncol] = (_Float16)y;
        }
    }
}

// ---------------------------------------------------------------------------
// Logits: (B*255,128) fp16 x (128,399) -> fp32.  W_nt (102 KB fp16) is staged
// in LDS ONCE per block (padded rows), then the block sweeps 16 row tiles,
// staging each 16x128 A tile in LDS.  Each wave holds its 4 A-fragments in
// registers and sweeps column tiles ct = wid, wid+8, ... (wave-uniform loop,
// EXEC stays full through the WMMAs; N=399 handled by clamped B rows and
// masked stores).
// ---------------------------------------------------------------------------
#define LOGITS_TPB 16   // row tiles per block; 16320/16 = 1020 blocks

__global__ __launch_bounds__(256)
void logits_kernel(const _Float16* __restrict__ codes,   // (B*255, 128)
                   const _Float16* __restrict__ Wnt,     // (399,128) f16
                   const float* __restrict__ bnt,        // (399)
                   float* __restrict__ out) {
    __shared__ _Float16 Ws[399 * NT_LDA];   // ~106 KB
    __shared__ _Float16 Asl[16 * NT_LDA];   // ~4.3 KB
    const int tid  = threadIdx.x;
    const int lane = tid & 31;
    const int wid  = tid >> 5;
    const int r    = lane & 15;
    const int kb   = (lane & 16) ? 8 : 0;
    const int hi8  = (lane & 16) ? 8 : 0;

    // stage full W_nt: 399 rows x 16 uint4 = 6384 vec4 copies
    for (int i = tid; i < 399 * 16; i += 256) {
        int rr = i >> 4, cc = i & 15;
        ((uint4*)(Ws + rr * NT_LDA))[cc] = ((const uint4*)(Wnt + (size_t)rr * 128))[cc];
    }

    for (int t = 0; t < LOGITS_TPB; ++t) {
        const int m0 = (blockIdx.x * LOGITS_TPB + t) * 16;
        __syncthreads();   // Ws ready (first iter) / previous As consumers done
        {
            int rr = tid >> 4, cc = tid & 15;
            ((uint4*)(Asl + rr * NT_LDA))[cc] =
                ((const uint4*)(codes + (size_t)(m0 + rr) * 128))[cc];
        }
        __syncthreads();

        const _Float16* ap = Asl + r * NT_LDA + kb;
        v16h a0 = make_frag(ap);
        v16h a1 = make_frag(ap + 32);
        v16h a2 = make_frag(ap + 64);
        v16h a3 = make_frag(ap + 96);

        for (int ct = wid; ct < 25; ct += 8) {   // wave-uniform
            int n  = ct * 16 + r;
            int nc = (n < 399) ? n : 398;
            const _Float16* wp = Ws + (size_t)nc * NT_LDA + kb;
            v8f acc = {};
            acc = WMMA(a0, make_frag(wp),      acc);
            acc = WMMA(a1, make_frag(wp + 32), acc);
            acc = WMMA(a2, make_frag(wp + 64), acc);
            acc = WMMA(a3, make_frag(wp + 96), acc);
            if (n < 399) {
                float bias = bnt[n];
#pragma unroll
                for (int v = 0; v < 8; ++v)
                    out[(size_t)(m0 + v + hi8) * 399 + n] = acc[v] + bias;
            }
        }
    }
}

// ---------------------------------------------------------------------------
// Host-side orchestration
// ---------------------------------------------------------------------------
extern "C" void kernel_launch(void* const* d_in, const int* in_sizes, int n_in,
                              void* d_out, int out_size, void* d_ws, size_t ws_size,
                              hipStream_t stream) {
    (void)in_sizes; (void)n_in; (void)out_size; (void)ws_size;

    const int*   leaf  = (const int*)  d_in[0];
    const int*   opt   = (const int*)  d_in[1];
    const float* eps   = (const float*)d_in[2];
    const float* emb   = (const float*)d_in[3];
    const float* Wenc  = (const float*)d_in[4];
    const float* benc  = (const float*)d_in[5];
    const float* Wmu   = (const float*)d_in[6];
    const float* bmu   = (const float*)d_in[7];
    const float* Wlv   = (const float*)d_in[8];
    const float* blv   = (const float*)d_in[9];
    const float* Wnt   = (const float*)d_in[10];
    const float* bnt   = (const float*)d_in[11];
    const float* Wdec  = (const float*)d_in[12];
    const float* bdec  = (const float*)d_in[13];

    float* out    = (float*)d_out;
    float* mu_out = out + (size_t)1024 * 255 * 399;
    float* lv_out = mu_out + 1024 * 128;

    _Float16* ws = (_Float16*)d_ws;
    size_t off = 0;
    _Float16* hbuf0  = ws + off; off += (size_t)1024 * 128 * 256;  // 64 MB
    _Float16* hbuf1  = ws + off; off += (size_t)1024 *  64 * 256;  // 32 MB
    _Float16* codes  = ws + off; off += (size_t)1024 * 255 * 128;  // 64 MB
    _Float16* wenc_h = ws + off; off += 3 * 256 * 512;
    _Float16* wdec_h = ws + off; off += 3 * 256 * 128;
    _Float16* wnt_h  = ws + off; off += 399 * 128;

    // weight conversion fp32 -> fp16
    f32_to_f16_kernel<<<(3 * 256 * 512 + 255) / 256, 256, 0, stream>>>(Wenc, wenc_h, 3 * 256 * 512);
    f32_to_f16_kernel<<<(3 * 256 * 128 + 255) / 256, 256, 0, stream>>>(Wdec, wdec_h, 3 * 256 * 128);
    f32_to_f16_kernel<<<(399 * 128   + 255) / 256, 256, 0, stream>>>(Wnt,  wnt_h,  399 * 128);

    // leaves
    embed_kernel<<<(1024 * 128 * 64) / 256, 256, 0, stream>>>(leaf, emb, hbuf0);

    // encoder: d = 6..0, ping-pong
    _Float16* cur = hbuf0;
    _Float16* nxt = hbuf1;
    for (int d = 6; d >= 0; --d) {
        int np = 1 << d;
        int M  = 1024 * np;
        enc_level_kernel<<<M / 32, 512, 0, stream>>>(cur, wenc_h, benc, opt, nxt, d, np - 1);
        _Float16* t = cur; cur = nxt; nxt = t;
    }

    // latent head: mu / logvar / z (z -> codes node 0)
    latent_kernel<<<(1024 * 128) / 256, 256, 0, stream>>>(cur, Wmu, bmu, Wlv, blv, eps,
                                                          mu_out, lv_out, codes);

    // decoder: d = 0..6, writes next level into codes in place
    for (int d = 0; d < 7; ++d) {
        int n = 1 << d;
        int M = 1024 * n;
        dec_level_kernel<<<M / 32, 512, 0, stream>>>(codes, wdec_h, bdec, opt, codes,
                                                     d, n - 1, 2 * n - 1, n - 1);
    }

    // logits over all 255 codes per batch: 16320 row tiles / 16 per block
    logits_kernel<<<16320 / LOGITS_TPB, 256, 0, stream>>>(codes, wnt_h, bnt, out);
}
